// Encoder_14508399526424
// MI455X (gfx1250) — matrix operand order repaired and checked by reading, not compiled, not run on hardware
//
#include <hip/hip_runtime.h>
#include <hip/hip_bf16.h>
#include <cstdint>

// ---------------- problem constants (fixed by the reference) ----------------
constexpr int NB    = 32;            // batches
constexpr int NPG   = 6890;          // nodes per graph
constexpr int NN    = NB * NPG;      // 220480 nodes  (== 13780 * 16, exact WMMA tiles)
constexpr int NE    = NN * 6;        // 1322880 edges
constexpr int MTILES = NN / 16;      // 13780

typedef __attribute__((ext_vector_type(16))) _Float16 v16h;
typedef __attribute__((ext_vector_type(8)))  float    v8f;

// ---------------- utility kernels ----------------
__global__ void zero_kernel(float* __restrict__ p, long n) {
  long stride = (long)gridDim.x * blockDim.x;
  for (long i = (long)blockIdx.x * blockDim.x + threadIdx.x; i < n; i += stride)
    p[i] = 0.0f;
}

__global__ void deg_kernel(const int* __restrict__ src, float* __restrict__ deg) {
  int e = blockIdx.x * blockDim.x + threadIdx.x;
  if (e < NE) atomicAdd(deg + src[e], 1.0f);
}

__global__ void dinv_kernel(const float* __restrict__ deg, float* __restrict__ dinv) {
  int i = blockIdx.x * blockDim.x + threadIdx.x;
  if (i < NN) {
    float d = deg[i];
    dinv[i] = (d > 0.0f) ? rsqrtf(fmaxf(d, 1.0f)) : 0.0f;
  }
}

__global__ void edgew_kernel(const int* __restrict__ src, const int* __restrict__ dst,
                             const float* __restrict__ dinv, float* __restrict__ w) {
  int e = blockIdx.x * blockDim.x + threadIdx.x;
  if (e < NE) w[e] = -dinv[src[e]] * dinv[dst[e]];
}

__global__ void h0_kernel(const float* __restrict__ x, const float* __restrict__ y,
                          const int* __restrict__ batch, float* __restrict__ h0) {
  int i = blockIdx.x * blockDim.x + threadIdx.x;
  if (i >= NN) return;
  int b = batch[i];
  h0[i * 6 + 0] = x[i * 3 + 0];
  h0[i * 6 + 1] = x[i * 3 + 1];
  h0[i * 6 + 2] = x[i * 3 + 2];
  h0[i * 6 + 3] = y[b * 3 + 0];
  h0[i * 6 + 4] = y[b * 3 + 1];
  h0[i * 6 + 5] = y[b * 3 + 2];
}

// ---------------- edge propagation: out[dst] += w * in[src] ----------------
__global__ void prop6_kernel(const int* __restrict__ src, const int* __restrict__ dst,
                             const float* __restrict__ w, const float* __restrict__ x,
                             float* __restrict__ out) {
  int e = blockIdx.x * blockDim.x + threadIdx.x;
  if (e >= NE) return;
  int s = src[e], d = dst[e];
  float we = w[e];
  #pragma unroll
  for (int c = 0; c < 6; ++c)
    atomicAdd(out + (long)d * 6 + c, we * x[(long)s * 6 + c]);
}

// 8 edges per block; 32 lanes per edge, 4 channels per lane (float4 row loads).
__global__ __launch_bounds__(256)
void prop128_kernel(const int* __restrict__ src, const int* __restrict__ dst,
                    const float* __restrict__ w, const float* __restrict__ x,
                    float* __restrict__ out) {
  long e = (long)blockIdx.x * 8 + (threadIdx.x >> 5);
  int  c = (threadIdx.x & 31) * 4;
  if (e >= NE) return;
  int s = src[e], d = dst[e];
  float we = w[e];
  const float4 v = *(const float4*)(x + (long)s * 128 + c);
  float* o = out + (long)d * 128 + c;
  atomicAdd(o + 0, we * v.x);
  atomicAdd(o + 1, we * v.y);
  atomicAdd(o + 2, we * v.z);
  atomicAdd(o + 3, we * v.w);
}

// ---------------- weight f32 -> f16, pre-swizzled into WMMA B-fragment order ----
// Output layout: [3 term][8 ntile][nchunks kchunk][32 lane][16 half]
// Fragment element i of lane L, tile nt, chunk kc:
//   n = nt*16 + (L&15), k = kc*32 + (L>>4)*8 + (i&7) + (i>>3)*16   (zero-padded for k >= cin)
__global__ void convw_swz_kernel(const float* __restrict__ wsrc, _Float16* __restrict__ wdst,
                                 int cin, int nchunks) {
  long total = 3L * 8 * nchunks * 32 * 16;
  long stride = (long)gridDim.x * blockDim.x;
  for (long idx = (long)blockIdx.x * blockDim.x + threadIdx.x; idx < total; idx += stride) {
    int i    = (int)(idx & 15);
    int lane = (int)((idx >> 4) & 31);
    long rem = idx >> 9;                 // term*8*nchunks + ntile*nchunks + chunk
    int chunk = (int)(rem % nchunks);
    long rem2 = rem / nchunks;
    int ntile = (int)(rem2 & 7);
    int term  = (int)(rem2 >> 3);
    int k = chunk * 32 + (lane >> 4) * 8 + (i & 7) + (i >> 3) * 16;
    int n = ntile * 16 + (lane & 15);
    float v = (k < cin) ? wsrc[((long)term * cin + k) * 128 + n] : 0.0f;
    wdst[idx] = (_Float16)v;
  }
}

// ---------------- fused Cheb GEMM:  out = Tx0@W0 + Tx1@W1 + (2*P2-Tx0)@W2 + b ----------------
// One block = 16 rows x 128 cols of output; 8 waves, each wave a 16x16 WMMA tile.
// `out` may alias `tx1` (each block reads its rows into LDS before writing).
template <int CINP>
__global__ __launch_bounds__(256)
void cheb_gemm_kernel(const float* __restrict__ tx0,
                      const float* __restrict__ tx1,
                      const float* __restrict__ p2,
                      const v16h* __restrict__ wfrag,    // swizzled [3][8][CINP/32][32] v16h
                      const float* __restrict__ bias,    // [128]
                      float* __restrict__ out,           // [NN][128]
                      int ncols_in) {
  constexpr int NCHUNK = CINP / 32;
  __shared__ _Float16 lds[3][16][CINP];
  const int tid = threadIdx.x;
  const int m0  = blockIdx.x * 16;

  // stage A operands (f32 -> f16), fusing Tx2 = 2*P2 - Tx0
  for (int idx = tid; idx < 16 * CINP; idx += 256) {
    int r = idx / CINP, c = idx - r * CINP;
    float t0 = 0.0f, t1 = 0.0f, t2 = 0.0f;
    if (c < ncols_in) {
      long g = (long)(m0 + r) * ncols_in + c;
      t0 = tx0[g];
      t1 = tx1[g];
      t2 = 2.0f * p2[g] - t0;
    }
    lds[0][r][c] = (_Float16)t0;
    lds[1][r][c] = (_Float16)t1;
    lds[2][r][c] = (_Float16)t2;
  }
  __syncthreads();

  const int wave = tid >> 5;           // 0..7 -> output column tile
  const int lane = tid & 31;
  const int arow = lane & 15;          // A: M = lane%16
  const int akb  = (lane >> 4) * 8;    // A: K half-split 0 / 8

  v8f acc = {0.f, 0.f, 0.f, 0.f, 0.f, 0.f, 0.f, 0.f};

  #pragma unroll
  for (int term = 0; term < 3; ++term) {
    #pragma unroll
    for (int kc = 0; kc < NCHUNK; ++kc) {
      v16h a;
      #pragma unroll
      for (int i = 0; i < 16; ++i) {
        // 16-bit A 16x32 layout: halves 0..7 -> K = akb+0..7, halves 8..15 -> K = akb+16..23
        int ka = (i < 8) ? (akb + i) : (akb + 8 + i);
        a[i] = lds[term][arow][kc * 32 + ka];
      }
      // B fragment: one aligned 32B load (pre-swizzled weights)
      v16h b = wfrag[(((long)term * 8 + wave) * NCHUNK + kc) * 32 + lane];
      acc = __builtin_amdgcn_wmma_f32_16x16x32_f16(false, a, false, b,
                                                   (short)0, acc, false, false);
    }
  }

  // D layout: VGPR r -> row = m0 + r + (lane<16 ? 0 : 8), col = n0 + lane%16
  const int col = wave * 16 + (lane & 15);
  const int rb  = (lane >> 4) * 8;
  const float bcol = bias[col];
  #pragma unroll
  for (int r = 0; r < 8; ++r)
    out[(long)(m0 + rb + r) * 128 + col] = acc[r] + bcol;
}

// ---------------- batchnorm (training-mode batch stats) + ReLU ----------------
__global__ __launch_bounds__(256)
void bn_partial_kernel(const float* __restrict__ x, float* __restrict__ sums,
                       int rows_per_block) {
  int c = threadIdx.x & 127;
  int half = threadIdx.x >> 7;
  int r0 = blockIdx.x * rows_per_block;
  int r1 = r0 + rows_per_block; if (r1 > NN) r1 = NN;
  float s = 0.f, s2 = 0.f;
  for (int i = r0 + half; i < r1; i += 2) {
    float v = x[(long)i * 128 + c];
    s += v; s2 += v * v;
  }
  __shared__ float sh[256], sh2[256];
  sh[threadIdx.x] = s; sh2[threadIdx.x] = s2;
  __syncthreads();
  if (half == 0) {
    atomicAdd(sums + c,       sh[c] + sh[c + 128]);
    atomicAdd(sums + 128 + c, sh2[c] + sh2[c + 128]);
  }
}

__global__ void bn_final_kernel(const float* __restrict__ sums, float* __restrict__ mv) {
  int c = threadIdx.x;
  if (c < 128) {
    float m = sums[c] / (float)NN;
    float v = sums[128 + c] / (float)NN - m * m;
    mv[c] = m; mv[128 + c] = v;
  }
}

__global__ void bn_relu_kernel(const float* __restrict__ x, float* __restrict__ out,
                               const float* __restrict__ mv, const float* __restrict__ g,
                               const float* __restrict__ be) {
  long total = (long)NN * 128;
  long stride = (long)gridDim.x * blockDim.x;
  for (long idx = (long)blockIdx.x * blockDim.x + threadIdx.x; idx < total; idx += stride) {
    int c = (int)(idx & 127);
    float m = mv[c], v = mv[128 + c];
    float val = (x[idx] - m) * rsqrtf(v + 1e-5f) * g[c] + be[c];
    out[idx] = fmaxf(val, 0.0f);
  }
}

// ---------------- mean pool per graph ----------------
constexpr int POOL_CHUNKS = 16;
__global__ __launch_bounds__(256)
void pool_partial_kernel(const float* __restrict__ x, float* __restrict__ pooled) {
  int b = blockIdx.x, chunk = blockIdx.y;
  int c = threadIdx.x & 127, half = threadIdx.x >> 7;
  int rows = (NPG + POOL_CHUNKS - 1) / POOL_CHUNKS;
  int r0 = chunk * rows;
  int r1 = r0 + rows; if (r1 > NPG) r1 = NPG;
  float s = 0.f;
  for (int i = r0 + half; i < r1; i += 2)
    s += x[((long)b * NPG + i) * 128 + c];
  __shared__ float sh[256];
  sh[threadIdx.x] = s;
  __syncthreads();
  if (half == 0) atomicAdd(pooled + b * 128 + c, sh[c] + sh[c + 128]);
}

// ---------------- VAE head: mu/logvar/z + three tiny MLPs ----------------
__global__ __launch_bounds__(32)
void head_kernel(const float* __restrict__ pooled, const float* __restrict__ eps,
                 const float* __restrict__ Wmu, const float* __restrict__ bmu,
                 const float* __restrict__ Wlv, const float* __restrict__ blv,
                 const float* __restrict__ Ww1, const float* __restrict__ bw1,
                 const float* __restrict__ Ww2, const float* __restrict__ bw2,
                 const float* __restrict__ Wh1, const float* __restrict__ bh1,
                 const float* __restrict__ Wh2, const float* __restrict__ bh2,
                 const float* __restrict__ Wg1, const float* __restrict__ bg1,
                 const float* __restrict__ Wg2, const float* __restrict__ bg2,
                 float* __restrict__ out) {
  int b = blockIdx.x, lane = threadIdx.x;
  __shared__ float muv[16], lvv[16], z[16], red[32];
  const float inv = 1.0f / (float)NPG;

  int j = lane & 15;
  const float* W  = (lane < 16) ? Wmu : Wlv;
  const float* bb = (lane < 16) ? bmu : blv;
  float val = 0.f;
  for (int c = 0; c < 128; ++c)
    val += pooled[b * 128 + c] * inv * W[c * 16 + j];
  val += bb[j];
  if (lane < 16) { muv[j] = val; out[b * 51 + j] = val; }
  else           { lvv[j] = val; out[b * 51 + 16 + j] = val; }
  __syncthreads();

  if (lane < 16) {
    float zz = muv[j] + eps[b * 16 + j] * expf(0.5f * lvv[j]);
    z[j] = zz;
    out[b * 51 + 32 + j] = zz;
  }
  __syncthreads();

  float hw = bw1[lane], hh = bh1[lane], hg = bg1[lane];
  #pragma unroll
  for (int i = 0; i < 6; ++i) hw += z[i] * Ww1[i * 32 + lane];
  #pragma unroll
  for (int i = 0; i < 6; ++i) hh += z[6 + i] * Wh1[i * 32 + lane];
  #pragma unroll
  for (int i = 0; i < 4; ++i) hg += z[12 + i] * Wg1[i * 32 + lane];
  hw = fmaxf(hw, 0.f) * Ww2[lane];
  hh = fmaxf(hh, 0.f) * Wh2[lane];
  hg = fmaxf(hg, 0.f) * Wg2[lane];

  red[lane] = hw; __syncthreads();
  if (lane == 0) { float s = bw2[0]; for (int i = 0; i < 32; ++i) s += red[i]; out[b * 51 + 48] = s; }
  __syncthreads();
  red[lane] = hh; __syncthreads();
  if (lane == 0) { float s = bh2[0]; for (int i = 0; i < 32; ++i) s += red[i]; out[b * 51 + 49] = s; }
  __syncthreads();
  red[lane] = hg; __syncthreads();
  if (lane == 0) { float s = bg2[0]; for (int i = 0; i < 32; ++i) s += red[i]; out[b * 51 + 50] = s; }
}

// ---------------- host orchestration ----------------
extern "C" void kernel_launch(void* const* d_in, const int* in_sizes, int n_in,
                              void* d_out, int out_size, void* d_ws, size_t ws_size,
                              hipStream_t stream) {
  const float* x   = (const float*)d_in[0];
  const float* y   = (const float*)d_in[1];
  const float* eps = (const float*)d_in[2];
  const float* W1c = (const float*)d_in[3];  const float* b1c = (const float*)d_in[4];
  const float* W2c = (const float*)d_in[5];  const float* b2c = (const float*)d_in[6];
  const float* W3c = (const float*)d_in[7];  const float* b3c = (const float*)d_in[8];
  const float* g1  = (const float*)d_in[9];  const float* be1 = (const float*)d_in[10];
  const float* g2  = (const float*)d_in[11]; const float* be2 = (const float*)d_in[12];
  const float* g3  = (const float*)d_in[13]; const float* be3 = (const float*)d_in[14];
  const float* Wmu = (const float*)d_in[15]; const float* bmu = (const float*)d_in[16];
  const float* Wlv = (const float*)d_in[17]; const float* blv = (const float*)d_in[18];
  const float* Ww1 = (const float*)d_in[19]; const float* bw1 = (const float*)d_in[20];
  const float* Ww2 = (const float*)d_in[21]; const float* bw2 = (const float*)d_in[22];
  const float* Wh1 = (const float*)d_in[23]; const float* bh1 = (const float*)d_in[24];
  const float* Wh2 = (const float*)d_in[25]; const float* bh2 = (const float*)d_in[26];
  const float* Wg1 = (const float*)d_in[27]; const float* bg1 = (const float*)d_in[28];
  const float* Wg2 = (const float*)d_in[29]; const float* bg2 = (const float*)d_in[30];
  const int* ei    = (const int*)d_in[31];
  const int* batch = (const int*)d_in[32];
  const int* src = ei;
  const int* dst = ei + NE;
  float* out = (float*)d_out;

  // workspace carve-up (floats; base is 256B aligned, all offsets multiples of 8 floats)
  float* ws = (float*)d_ws;
  long off = 0;
  float* deg  = ws + off; off += NN;
  float* dinv = ws + off; off += NN;
  float* ew   = ws + off; off += NE;
  float* h0   = ws + off; off += (long)NN * 6;
  float* A    = ws + off; off += (long)NN * 128;   // activations / layer-1 GEMM out
  float* U    = ws + off; off += (long)NN * 128;   // Tx1 / GEMM out (aliased)
  float* V    = ws + off; off += (long)NN * 128;   // P2 = prop(Tx1)
  v16h* wf1 = (v16h*)(ws + off); off += (3L * 32 * 128) / 2;    // layer1: 1 chunk
  v16h* wf2 = (v16h*)(ws + off); off += (3L * 128 * 128) / 2;   // 4 chunks
  v16h* wf3 = (v16h*)(ws + off); off += (3L * 128 * 128) / 2;
  float* bnsum  = ws + off; off += 256;
  float* bnmv   = ws + off; off += 256;
  float* pooled = ws + off; off += (long)NB * 128;

  const int ZB = 2048;
  const int egrid = (NE + 255) / 256;
  const int ngrid = (NN + 255) / 256;
  const int pgrid = (NE + 7) / 8;               // prop128: 8 edges per block
  const int bn_rows = 256;
  const int bn_blocks = (NN + bn_rows - 1) / bn_rows;

  // graph normalization
  zero_kernel<<<ZB, 256, 0, stream>>>(deg, NN);
  deg_kernel<<<egrid, 256, 0, stream>>>(src, deg);
  dinv_kernel<<<ngrid, 256, 0, stream>>>(deg, dinv);
  edgew_kernel<<<egrid, 256, 0, stream>>>(src, dst, dinv, ew);

  // input features [N,6]
  h0_kernel<<<ngrid, 256, 0, stream>>>(x, y, batch, h0);

  // weight conversion f32 -> f16, swizzled to WMMA fragment order
  convw_swz_kernel<<<64, 256, 0, stream>>>(W1c, (_Float16*)wf1, 6, 1);
  convw_swz_kernel<<<256, 256, 0, stream>>>(W2c, (_Float16*)wf2, 128, 4);
  convw_swz_kernel<<<256, 256, 0, stream>>>(W3c, (_Float16*)wf3, 128, 4);

  // ---- layer 1 (Cin=6) ----
  zero_kernel<<<ZB, 256, 0, stream>>>(U, (long)NN * 6);
  prop6_kernel<<<egrid, 256, 0, stream>>>(src, dst, ew, h0, U);       // Tx1
  zero_kernel<<<ZB, 256, 0, stream>>>(V, (long)NN * 6);
  prop6_kernel<<<egrid, 256, 0, stream>>>(src, dst, ew, U, V);        // P2
  cheb_gemm_kernel<32><<<MTILES, 256, 0, stream>>>(h0, U, V, wf1, b1c, A, 6);
  zero_kernel<<<4, 256, 0, stream>>>(bnsum, 256);
  bn_partial_kernel<<<bn_blocks, 256, 0, stream>>>(A, bnsum, bn_rows);
  bn_final_kernel<<<1, 128, 0, stream>>>(bnsum, bnmv);
  bn_relu_kernel<<<ZB, 256, 0, stream>>>(A, A, bnmv, g1, be1);

  // ---- layer 2 (Cin=128) ----
  zero_kernel<<<ZB, 256, 0, stream>>>(U, (long)NN * 128);
  prop128_kernel<<<pgrid, 256, 0, stream>>>(src, dst, ew, A, U);
  zero_kernel<<<ZB, 256, 0, stream>>>(V, (long)NN * 128);
  prop128_kernel<<<pgrid, 256, 0, stream>>>(src, dst, ew, U, V);
  cheb_gemm_kernel<128><<<MTILES, 256, 0, stream>>>(A, U, V, wf2, b2c, U, 128); // out aliases Tx1
  zero_kernel<<<4, 256, 0, stream>>>(bnsum, 256);
  bn_partial_kernel<<<bn_blocks, 256, 0, stream>>>(U, bnsum, bn_rows);
  bn_final_kernel<<<1, 128, 0, stream>>>(bnsum, bnmv);
  bn_relu_kernel<<<ZB, 256, 0, stream>>>(U, A, bnmv, g2, be2);

  // ---- layer 3 (Cin=128) ----
  zero_kernel<<<ZB, 256, 0, stream>>>(U, (long)NN * 128);
  prop128_kernel<<<pgrid, 256, 0, stream>>>(src, dst, ew, A, U);
  zero_kernel<<<ZB, 256, 0, stream>>>(V, (long)NN * 128);
  prop128_kernel<<<pgrid, 256, 0, stream>>>(src, dst, ew, U, V);
  cheb_gemm_kernel<128><<<MTILES, 256, 0, stream>>>(A, U, V, wf3, b3c, U, 128);
  zero_kernel<<<4, 256, 0, stream>>>(bnsum, 256);
  bn_partial_kernel<<<bn_blocks, 256, 0, stream>>>(U, bnsum, bn_rows);
  bn_final_kernel<<<1, 128, 0, stream>>>(bnsum, bnmv);
  bn_relu_kernel<<<ZB, 256, 0, stream>>>(U, A, bnmv, g3, be3);

  // ---- pool + head ----
  zero_kernel<<<16, 256, 0, stream>>>(pooled, (long)NB * 128);
  pool_partial_kernel<<<dim3(NB, POOL_CHUNKS), 256, 0, stream>>>(A, pooled);
  head_kernel<<<NB, 32, 0, stream>>>(pooled, eps, Wmu, bmu, Wlv, blv,
                                     Ww1, bw1, Ww2, bw2, Wh1, bh1, Wh2, bh2,
                                     Wg1, bg1, Wg2, bg2, out);
}
